// FeatureHead_39178691674640
// MI455X (gfx1250) — matrix operand order repaired and checked by reading, not compiled, not run on hardware
//
#include <hip/hip_runtime.h>
#include <cstdint>

// ---------------------------------------------------------------------------
// CDNA5 (gfx1250) SwinV2 feature head.
// GEMMs on v_wmma_f32_16x16x32_bf16; double-buffered async global->LDS tile
// pipeline (global_load_async_to_lds_b128 + s_wait_asynccnt), MT=64 so each
// wave runs 8 WMMAs per barrier pair; coalesced 128-bit epilogue stores
// staged through LDS in two 32-row passes.
// ---------------------------------------------------------------------------

typedef __bf16 v16bf __attribute__((ext_vector_type(16)));
typedef float  v8f   __attribute__((ext_vector_type(8)));

#define DEVINL __device__ __forceinline__

DEVINL uint32_t lds_off(const void* p) {
  // addrspace(3) offsets live in the low 32 bits of the generic address
  return (uint32_t)(uintptr_t)p;
}

DEVINL void cp_glb_to_lds16(uint32_t lds_byte, const void* g) {
  asm volatile("global_load_async_to_lds_b128 %0, %1, off"
               :: "v"(lds_byte), "v"((unsigned long long)(uintptr_t)g)
               : "memory");
}
DEVINL void wait_async0() { asm volatile("s_wait_asynccnt 0x0" ::: "memory"); }
DEVINL void wait_async5() { asm volatile("s_wait_asynccnt 0x5" ::: "memory"); }

DEVINL v8f wmma_bf16(v16bf a, v16bf b, v8f c) {
  return __builtin_amdgcn_wmma_f32_16x16x32_bf16(
      false, a, false, b, (short)0, c, false, false);
}

DEVINL uint32_t pack2(float lo, float hi) {
  union { __bf16 h[2]; uint32_t u; } t;
  t.h[0] = (__bf16)lo;
  t.h[1] = (__bf16)hi;
  return t.u;
}

// A fragment: 16x32 (MxK) tile, LDS row-major [row][k], ld in elements.
DEVINL v16bf frag_a(const __bf16* base, int ld, int row0, int k0) {
  const int lane = threadIdx.x & 31;
  const __bf16* p = base + (row0 + (lane & 15)) * ld + k0;
  const int kb = (lane >> 4) << 4;
  v16bf a;
#pragma unroll
  for (int i = 0; i < 8; ++i) {
    int k = kb + 2 * i;
    a[2 * i]     = p[k];
    a[2 * i + 1] = p[k + 1];
  }
  return a;
}

// B fragment: 32x16 (KxN) tile stored transposed in LDS as Bt[n][k].
DEVINL v16bf frag_b(const __bf16* base, int ld, int col0, int k0) {
  const int lane = threadIdx.x & 31;
  const __bf16* p = base + (col0 + (lane & 15)) * ld + k0 + ((lane >> 4) << 4);
  v16bf b;
#pragma unroll
  for (int i = 0; i < 8; ++i) {
    b[2 * i]     = p[2 * i];
    b[2 * i + 1] = p[2 * i + 1];
  }
  return b;
}

// ---------------------------------------------------------------------------
// f32 -> bf16 conversion (weights)
// ---------------------------------------------------------------------------
__global__ __launch_bounds__(256) void cvt_bf16_kernel(
    const float* __restrict__ in, __bf16* __restrict__ out, int n) {
  for (int i = blockIdx.x * 256 + threadIdx.x; i < n; i += gridDim.x * 256)
    out[i] = (__bf16)in[i];
}

// ---------------------------------------------------------------------------
// Continuous position bias: bias[head][n][m], head<8, n,m<64
// ---------------------------------------------------------------------------
__global__ __launch_bounds__(256) void cpb_kernel(
    const float* __restrict__ w1, const float* __restrict__ b1,
    const float* __restrict__ w2, float* __restrict__ biasout) {
  __shared__ float tab[225][8];
  const int tid = threadIdx.x;
  if (tid < 225) {
    int i = tid / 15, j = tid % 15;
    float t0 = (float)(i - 7) * (8.0f / 7.0f);
    float t1 = (float)(j - 7) * (8.0f / 7.0f);
    t0 = (t0 >= 0.f ? 1.f : -1.f) * log2f(fabsf(t0) + 1.f) * (1.0f / 3.0f);
    t1 = (t1 >= 0.f ? 1.f : -1.f) * log2f(fabsf(t1) + 1.f) * (1.0f / 3.0f);
    float acc[8] = {0, 0, 0, 0, 0, 0, 0, 0};
    for (int jj = 0; jj < 512; ++jj) {
      float h = fmaxf(t0 * w1[jj * 2 + 0] + t1 * w1[jj * 2 + 1] + b1[jj], 0.f);
#pragma unroll
      for (int hh = 0; hh < 8; ++hh) acc[hh] += h * w2[hh * 512 + jj];
    }
#pragma unroll
    for (int hh = 0; hh < 8; ++hh)
      tab[tid][hh] = 16.0f / (1.0f + expf(-acc[hh]));
  }
  __syncthreads();
  for (int e = tid; e < 8 * 64 * 64; e += 256) {
    int head = e >> 12, n = (e >> 6) & 63, m = e & 63;
    int idx = ((n >> 3) - (m >> 3) + 7) * 15 + ((n & 7) - (m & 7) + 7);
    biasout[e] = tab[idx][head];
  }
}

// ---------------------------------------------------------------------------
// Generic bf16 WMMA GEMM: out[M][N] = A[M][K] @ W[N][K]^T + bias, epilogues.
// Block 256 threads (8 waves). MT=64 rows, NT=256 cols per workgroup.
// Non-XSRC path: double-buffered async-to-LDS pipeline (5 asyncs/wave/stage).
// ---------------------------------------------------------------------------
enum { EPI_STOREBOTH = 0, EPI_BF16 = 1, EPI_GELU = 2, EPI_LNRES = 3 };

template <int EPI, bool XSRC>
__global__ __launch_bounds__(256) void gemm_kernel(
    const void* __restrict__ Asrc, const __bf16* __restrict__ Wt,
    const float* __restrict__ bias, float* __restrict__ outF,
    __bf16* __restrict__ outB, const float* __restrict__ resid,
    const float* __restrict__ lng, const float* __restrict__ lnb,
    int N, int K) {
  // Arena: [tiles: At[2][64][40] (10240B) | Bt[2][256][40] (40960B)] overlaid
  // with [epilogue: outS[32][264] (33792B) | red[32][16] (2048B)].
  __shared__ __align__(16) unsigned char smem[51200];
  __bf16* AtP = (__bf16*)smem;                   // [2][64][40]
  __bf16* BtP = (__bf16*)(smem + 10240);         // [2][256][40]
  float*  outS = (float*)smem;                   // [32][264]
  float*  red  = (float*)(smem + 33792);         // [32][16]

  const int tid = threadIdx.x;
  const int wv = tid >> 5;
  const int lane = tid & 31;
  const int mblk = blockIdx.x;
  const int n_off = blockIdx.y * 256;

  v8f acc[4][2];
#pragma unroll
  for (int mt = 0; mt < 4; ++mt)
#pragma unroll
    for (int nt = 0; nt < 2; ++nt) acc[mt][nt] = (v8f){};

  auto issueB = [&](int k0, int buf) {
    const __bf16* src = Wt + (long)(n_off + tid) * K + k0;
    uint32_t d = lds_off(&BtP[(buf * 256 + tid) * 40]);
#pragma unroll
    for (int s = 0; s < 4; ++s) cp_glb_to_lds16(d + s * 16, src + s * 8);
  };
  auto issueA = [&](int k0, int buf) {
    // 64 rows x 32 cols bf16 = 4KB: one b128 per thread
    const __bf16* src = (const __bf16*)Asrc +
                        (long)(mblk * 64 + (tid >> 2)) * K + k0 + (tid & 3) * 8;
    cp_glb_to_lds16(lds_off(&AtP[(buf * 64 + (tid >> 2)) * 40 + (tid & 3) * 8]),
                    src);
  };
  auto compute = [&](int buf) {
    const __bf16* ab = &AtP[buf * 64 * 40];
    const __bf16* bb = &BtP[buf * 256 * 40];
    v16bf b0 = frag_b(bb, 40, wv * 32, 0);
    v16bf b1 = frag_b(bb, 40, wv * 32 + 16, 0);
#pragma unroll
    for (int mt = 0; mt < 4; ++mt) {
      v16bf a = frag_a(ab, 40, mt * 16, 0);
      acc[mt][0] = wmma_bf16(a, b0, acc[mt][0]);
      acc[mt][1] = wmma_bf16(a, b1, acc[mt][1]);
    }
  };

  if (XSRC) {
    // A streamed from channel-major f32 x with in-LDS transpose+convert.
    for (int k0 = 0; k0 < K; k0 += 32) {
      issueB(k0, 0);
      {
        const float* xf = (const float*)Asrc;
        int c = tid >> 3, j = tid & 7;
        long bidx = (long)(mblk * 64) >> 12;
        long hw0  = (long)(mblk * 64) & 4095;
        const float* p = xf + (((bidx * 2048) + k0 + c) << 12) + hw0 + j * 8;
        const float4 v0 = *(const float4*)p;
        const float4 v1 = *(const float4*)(p + 4);
        AtP[(j * 8 + 0) * 40 + c] = (__bf16)v0.x;
        AtP[(j * 8 + 1) * 40 + c] = (__bf16)v0.y;
        AtP[(j * 8 + 2) * 40 + c] = (__bf16)v0.z;
        AtP[(j * 8 + 3) * 40 + c] = (__bf16)v0.w;
        AtP[(j * 8 + 4) * 40 + c] = (__bf16)v1.x;
        AtP[(j * 8 + 5) * 40 + c] = (__bf16)v1.y;
        AtP[(j * 8 + 6) * 40 + c] = (__bf16)v1.z;
        AtP[(j * 8 + 7) * 40 + c] = (__bf16)v1.w;
      }
      wait_async0();
      __syncthreads();
      compute(0);
      __syncthreads();
    }
  } else {
    issueB(0, 0);
    issueA(0, 0);
    for (int k0 = 0; k0 < K; k0 += 32) {
      const int cur = (k0 >> 5) & 1;
      if (k0 + 32 < K) {
        issueB(k0 + 32, cur ^ 1);
        issueA(k0 + 32, cur ^ 1);
        wait_async5();  // retire current stage, keep next in flight
      } else {
        wait_async0();
      }
      __syncthreads();
      compute(cur);
      __syncthreads();
    }
  }

  // ---- epilogue: two 32-row passes staged to LDS, 128-bit stores ----
  const int r16 = lane >> 4, c16 = lane & 15;
  const float bc0 = bias[n_off + wv * 32 + c16];
  const float bc1 = bias[n_off + wv * 32 + 16 + c16];

#pragma unroll
  for (int p = 0; p < 2; ++p) {
#pragma unroll
    for (int mt2 = 0; mt2 < 2; ++mt2) {
      const int mt = p * 2 + mt2;
#pragma unroll
      for (int nt = 0; nt < 2; ++nt) {
        const float bc = (nt == 0) ? bc0 : bc1;
#pragma unroll
        for (int i = 0; i < 8; ++i) {
          int row = mt2 * 16 + i + 8 * r16;  // 0..31 within pass
          int col = wv * 32 + nt * 16 + c16;
          outS[row * 264 + col] = acc[mt][nt][i] + bc;
        }
      }
    }
    __syncthreads();

    const int row = tid >> 3, seg = tid & 7;
    float vals[32];
#pragma unroll
    for (int j = 0; j < 32; ++j) vals[j] = outS[row * 264 + seg * 32 + j];

    if (EPI == EPI_GELU) {
#pragma unroll
      for (int j = 0; j < 32; ++j) {
        float v = vals[j];
        vals[j] = 0.5f * v * (1.0f + erff(v * 0.70710678f));
      }
    }

    const long grow = (long)mblk * 64 + p * 32 + row;
    const long gcol = (long)n_off + seg * 32;

    if (EPI == EPI_LNRES) {  // N == 256: full row in this workgroup
      float s = 0.f, s2 = 0.f;
#pragma unroll
      for (int j = 0; j < 32; ++j) { s += vals[j]; s2 += vals[j] * vals[j]; }
      red[row * 16 + seg] = s;
      red[row * 16 + 8 + seg] = s2;
      __syncthreads();
      if (tid < 32) {
        float ts = 0.f, ts2 = 0.f;
#pragma unroll
        for (int j = 0; j < 8; ++j) {
          ts += red[tid * 16 + j];
          ts2 += red[tid * 16 + 8 + j];
        }
        float mean = ts * (1.0f / 256.0f);
        float var = ts2 * (1.0f / 256.0f) - mean * mean;
        red[tid * 16 + 0] = mean;
        red[tid * 16 + 1] = rsqrtf(var + 1e-5f);
      }
      __syncthreads();
      const float mean = red[row * 16 + 0], rstd = red[row * 16 + 1];
#pragma unroll
      for (int j = 0; j < 32; ++j) {
        int col = seg * 32 + j;
        float nv = (vals[j] - mean) * rstd * lng[col] + lnb[col];
        vals[j] = resid[grow * 256 + col] + nv;
      }
    }

    if (EPI == EPI_STOREBOTH || EPI == EPI_LNRES) {
#pragma unroll
      for (int q = 0; q < 8; ++q) {
        float4 f;
        f.x = vals[q * 4 + 0]; f.y = vals[q * 4 + 1];
        f.z = vals[q * 4 + 2]; f.w = vals[q * 4 + 3];
        *(float4*)(outF + grow * N + gcol + q * 4) = f;
      }
    }
#pragma unroll
    for (int q = 0; q < 4; ++q) {
      uint4 w;
      w.x = pack2(vals[q * 8 + 0], vals[q * 8 + 1]);
      w.y = pack2(vals[q * 8 + 2], vals[q * 8 + 3]);
      w.z = pack2(vals[q * 8 + 4], vals[q * 8 + 5]);
      w.w = pack2(vals[q * 8 + 6], vals[q * 8 + 7]);
      *(uint4*)(outB + grow * N + gcol + q * 8) = w;
    }
    __syncthreads();  // outS reused by next pass
  }
}

// ---------------------------------------------------------------------------
// Window attention (SwinV2 cosine attention). Block = 64 threads = 2 waves,
// one head per wave. grid = (512 windows, 4 head pairs).
// ---------------------------------------------------------------------------
DEVINL int region_of(int t) { return t < 56 ? 0 : (t < 60 ? 1 : 2); }

__global__ __launch_bounds__(64) void attn_kernel(
    const __bf16* __restrict__ qkv,    // [32768][768]
    __bf16* __restrict__ ctx,          // [32768][256]
    const float* __restrict__ biasbuf, // [8][64][64]
    const float* __restrict__ lscale,  // [8]
    int shift) {
  __shared__ __bf16 qn[2][64][36];
  __shared__ __bf16 kn[2][64][36];
  __shared__ __bf16 vT[2][32][72];
  __shared__ float  Sb[2][16][68];
  __shared__ __bf16 Pb[2][16][72];
  __shared__ __align__(16) __bf16 Ob[2][16][32];

  const int wave = threadIdx.x >> 5;
  const int lane = threadIdx.x & 31;
  const int head = blockIdx.y * 2 + wave;
  const int win = blockIdx.x;
  const int b = win >> 6;
  const int wy = (win >> 3) & 7;
  const int wx = win & 7;

  const float scale = expf(fminf(lscale[head], 4.6051702f));

  // gather (vectorized 64B loads) + cosine normalize q,k; stash v transposed
  for (int t = lane; t < 64; t += 32) {
    int li = t >> 3, lj = t & 7;
    int sh = (wy * 8 + li + shift) & 63;
    int sw = (wx * 8 + lj + shift) & 63;
    long trow = (long)(b * 4096 + sh * 64 + sw);
    const __bf16* qp = qkv + trow * 768 + head * 32;
    union Pack { uint4 u[4]; __bf16 h[32]; } Q, Kk, V;
#pragma unroll
    for (int s = 0; s < 4; ++s) {
      Q.u[s]  = ((const uint4*)qp)[s];
      Kk.u[s] = ((const uint4*)(qp + 256))[s];
      V.u[s]  = ((const uint4*)(qp + 512))[s];
    }
    float qs = 0.f, ks = 0.f;
#pragma unroll
    for (int c = 0; c < 32; ++c) {
      float q = (float)Q.h[c], k = (float)Kk.h[c];
      qs += q * q; ks += k * k;
    }
    float qr = rsqrtf(fmaxf(qs, 1e-24f));
    float kr = rsqrtf(fmaxf(ks, 1e-24f));
#pragma unroll
    for (int c = 0; c < 32; ++c) {
      qn[wave][t][c] = (__bf16)((float)Q.h[c] * qr);
      kn[wave][t][c] = (__bf16)((float)Kk.h[c] * kr);
      vT[wave][c][t] = V.h[c];
    }
  }
  __syncthreads();

  const int r16 = lane >> 4, c16 = lane & 15;

  for (int mt = 0; mt < 4; ++mt) {
    // S block (16 queries x 64 keys)
    v16bf a = frag_a(&qn[wave][0][0], 36, mt * 16, 0);
#pragma unroll
    for (int nt = 0; nt < 4; ++nt) {
      v16bf bf = frag_b(&kn[wave][0][0], 36, nt * 16, 0);
      v8f sacc = {};
      sacc = wmma_bf16(a, bf, sacc);
#pragma unroll
      for (int i = 0; i < 8; ++i) {
        int row = mt * 16 + i + 8 * r16;  // query token in window
        int col = nt * 16 + c16;          // key token
        float s = sacc[i] * scale + biasbuf[head * 4096 + row * 64 + col];
        if (shift) {
          int lr = 3 * region_of(wy * 8 + (row >> 3)) + region_of(wx * 8 + (row & 7));
          int lc = 3 * region_of(wy * 8 + (col >> 3)) + region_of(wx * 8 + (col & 7));
          if (lr != lc) s -= 100.f;
        }
        Sb[wave][row - mt * 16][col] = s;
      }
    }
    // softmax: 2 lanes per row, 32 cols each
    {
      const int srow = lane >> 1, half = lane & 1;
      float m = -1e30f;
#pragma unroll
      for (int j = 0; j < 32; ++j) m = fmaxf(m, Sb[wave][srow][half * 32 + j]);
      m = fmaxf(m, __shfl_xor(m, 1, 32));
      float e[32], ssum = 0.f;
#pragma unroll
      for (int j = 0; j < 32; ++j) {
        e[j] = __expf(Sb[wave][srow][half * 32 + j] - m);
        ssum += e[j];
      }
      ssum += __shfl_xor(ssum, 1, 32);
      float inv = 1.0f / ssum;
#pragma unroll
      for (int j = 0; j < 32; ++j)
        Pb[wave][srow][half * 32 + j] = (__bf16)(e[j] * inv);
    }
    // O block = P(16x64) @ V(64x32), staged to LDS then 128-bit stores
    v16bf pa0 = frag_a(&Pb[wave][0][0], 72, 0, 0);
    v16bf pa1 = frag_a(&Pb[wave][0][0], 72, 0, 32);
#pragma unroll
    for (int nt2 = 0; nt2 < 2; ++nt2) {
      v8f oacc = {};
      oacc = wmma_bf16(pa0, frag_b(&vT[wave][0][0], 72, nt2 * 16, 0), oacc);
      oacc = wmma_bf16(pa1, frag_b(&vT[wave][0][0], 72, nt2 * 16, 32), oacc);
#pragma unroll
      for (int i = 0; i < 8; ++i)
        Ob[wave][i + 8 * r16][nt2 * 16 + c16] = (__bf16)oacc[i];
    }
    {
      const int orow = lane >> 1, ohalf = lane & 1;
      const int wrow = mt * 16 + orow;
      int li = wrow >> 3, lj = wrow & 7;
      int sh = (wy * 8 + li + shift) & 63;
      int sw = (wx * 8 + lj + shift) & 63;
      long trow = (long)(b * 4096 + sh * 64 + sw);
      uint4 u0 = *(const uint4*)&Ob[wave][orow][ohalf * 16];
      uint4 u1 = *(const uint4*)&Ob[wave][orow][ohalf * 16 + 8];
      *(uint4*)(ctx + trow * 256 + head * 32 + ohalf * 16) = u0;
      *(uint4*)(ctx + trow * 256 + head * 32 + ohalf * 16 + 8) = u1;
    }
  }
}

// ---------------------------------------------------------------------------
// Host driver
// ---------------------------------------------------------------------------
extern "C" void kernel_launch(void* const* d_in, const int* in_sizes, int n_in,
                              void* d_out, int out_size, void* d_ws,
                              size_t ws_size, hipStream_t stream) {
  (void)in_sizes; (void)n_in; (void)out_size; (void)ws_size;

  const float* x       = (const float*)d_in[0];
  const float* align_w = (const float*)d_in[1];
  const float* align_b = (const float*)d_in[2];
  auto P = [&](int blk, int j) -> const float* {
    return (const float*)d_in[3 + blk * 16 + j];
  };
  // block param order: 0 qkv_w 1 qkv_b 2 proj_w 3 proj_b 4 logit_scale
  // 5 cpb_w1 6 cpb_b1 7 cpb_w2 8 ln1_g 9 ln1_b 10 mlp_w1 11 mlp_b1
  // 12 mlp_w2 13 mlp_b2 14 ln2_g 15 ln2_b

  char* base = (char*)d_ws;
  size_t off = 0;
  auto carve = [&](size_t bytes) -> void* {
    void* p = base + off;
    off += (bytes + 255) & ~(size_t)255;
    return p;
  };
  float*  y     = (float*)carve(33554432);   // (32768, 256) f32
  __bf16* ybf   = (__bf16*)carve(16777216);  // bf16 mirror of y
  __bf16* qkvb  = (__bf16*)carve(50331648);  // (32768, 768) bf16
  __bf16* ctx   = (__bf16*)carve(16777216);  // (32768, 256) bf16
  __bf16* hbuf  = (__bf16*)carve(67108864);  // (32768, 1024) bf16
  float*  biasb = (float*)carve(131072);     // (8, 64, 64) f32
  __bf16* wA    = (__bf16*)carve(1048576);   // align_w bf16
  __bf16* wqkv  = (__bf16*)carve(393216);
  __bf16* wproj = (__bf16*)carve(131072);
  __bf16* w1b   = (__bf16*)carve(524288);
  __bf16* w2b   = (__bf16*)carve(524288);

  // align: y = x(1x1 conv); A tile comes straight from channel-major x
  cvt_bf16_kernel<<<256, 256, 0, stream>>>(align_w, wA, 256 * 2048);
  gemm_kernel<EPI_STOREBOTH, true><<<dim3(512, 1), 256, 0, stream>>>(
      x, wA, align_b, y, ybf, nullptr, nullptr, nullptr, 256, 2048);

  for (int blk = 0; blk < 6; ++blk) {
    const int shift = (blk & 1) ? 4 : 0;
    cvt_bf16_kernel<<<256, 256, 0, stream>>>(P(blk, 0), wqkv, 768 * 256);
    cvt_bf16_kernel<<<256, 256, 0, stream>>>(P(blk, 2), wproj, 256 * 256);
    cvt_bf16_kernel<<<256, 256, 0, stream>>>(P(blk, 10), w1b, 1024 * 256);
    cvt_bf16_kernel<<<256, 256, 0, stream>>>(P(blk, 12), w2b, 256 * 1024);
    cpb_kernel<<<1, 256, 0, stream>>>(P(blk, 5), P(blk, 6), P(blk, 7), biasb);

    gemm_kernel<EPI_BF16, false><<<dim3(512, 3), 256, 0, stream>>>(
        ybf, wqkv, P(blk, 1), nullptr, qkvb, nullptr, nullptr, nullptr,
        768, 256);
    attn_kernel<<<dim3(512, 4), 64, 0, stream>>>(qkvb, ctx, biasb, P(blk, 4),
                                                 shift);
    gemm_kernel<EPI_LNRES, false><<<dim3(512, 1), 256, 0, stream>>>(
        ctx, wproj, P(blk, 3), y, ybf, y, P(blk, 8), P(blk, 9), 256, 256);
    gemm_kernel<EPI_GELU, false><<<dim3(512, 4), 256, 0, stream>>>(
        ybf, w1b, P(blk, 11), nullptr, hbuf, nullptr, nullptr, nullptr,
        1024, 256);
    gemm_kernel<EPI_LNRES, false><<<dim3(512, 1), 256, 0, stream>>>(
        hbuf, w2b, P(blk, 13), y, ybf, y, P(blk, 14), P(blk, 15), 256, 1024);
  }

  hipMemcpyAsync(d_out, y, 33554432, hipMemcpyDeviceToDevice, stream);
}
